// Attention_13632226198096
// MI455X (gfx1250) — matrix operand order repaired and checked by reading, not compiled, not run on hardware
//
#include <hip/hip_runtime.h>

// ---------------------------------------------------------------------------
// Types for CDNA5 WMMA (wave32, 16x16x32 bf16 -> f32 accum)
// ---------------------------------------------------------------------------
typedef __attribute__((ext_vector_type(16))) __bf16 v16bf;
typedef __attribute__((ext_vector_type(8)))  __bf16 v8bf;
typedef __attribute__((ext_vector_type(8)))  float  v8f;
typedef __attribute__((ext_vector_type(4)))  unsigned int v4u;
typedef __attribute__((ext_vector_type(8)))  unsigned int v8u;

#define TILE_M 256   // block tile rows
#define TILE_N 128   // block tile cols
#define TILE_K 32    // K step (one bf16 WMMA K)
#define LDT    40    // padded LDS row stride in bf16 elems (64B data + 16B pad = 80B)

// ---------------------------------------------------------------------------
// Tensor Data Mover descriptors (CDNA5 D#, CDNA5 ISA)
// Tile: `rows` x 32 bf16 (64B) per row, memory row stride Kd elems.
// TDM pad: 4 DWORDs (16B) after every 16 DWORDs (64B) -> LDS row stride 80B,
// exactly the LDT=40 layout the compute side reads.
// ---------------------------------------------------------------------------
__device__ __forceinline__ v8u tdm_group1(int Kd, int rows) {
  v8u g;
  g[0] = (1u << 16)        // data_size = 1 -> 2 bytes
       | (1u << 20)        // pad_enable
       | (3u << 22)        // pad_interval code 3 -> every 16 DWORDs (64B)
       | (3u << 25);       // pad_amount  code 3 -> 4 DWORDs (16B)
  g[1] = ((unsigned)Kd & 0xFFFFu) << 16;                   // tensor_dim0 lo16 @[63:48]
  g[2] = ((unsigned)Kd >> 16) | ((unsigned)rows << 16);    // tensor_dim0 hi16, tensor_dim1 lo16
  g[3] = (32u << 16);                                      // tensor_dim1 hi16=0, tile_dim0=32
  g[4] = (unsigned)rows;                                   // tile_dim1=rows, tile_dim2=0
  g[5] = (unsigned)Kd;                                     // tensor_dim0_stride lo32
  g[6] = 0u;
  g[7] = 0u;
  return g;
}

__device__ __forceinline__ void tdm_load_tile(const __bf16* gptr, unsigned lds_byte_addr,
                                              v8u g1) {
  unsigned long long ga = (unsigned long long)(size_t)gptr;
  v4u g0;
  g0[0] = 1u;                                                  // count = 1 valid descriptor
  g0[1] = lds_byte_addr;                                       // lds_addr @ bits[63:32]
  g0[2] = (unsigned)ga;                                        // global_addr lo32
  g0[3] = (unsigned)((ga >> 32) & 0x01FFFFFFu) | (2u << 30);   // addr hi25, type=2
  asm volatile("tensor_load_to_lds %0, %1" :: "s"(g0), "s"(g1) : "memory");
}

// ---------------------------------------------------------------------------
// fp32 -> bf16 bulk convert (n must be a multiple of 2048)
// ---------------------------------------------------------------------------
__global__ __launch_bounds__(256)
void f32_to_bf16_kernel(const float* __restrict__ src, __bf16* __restrict__ dst) {
  long long i = ((long long)blockIdx.x * 256 + threadIdx.x) * 8;
  float4 a = *(const float4*)(src + i);
  float4 b = *(const float4*)(src + i + 4);
  v8bf o;
  o[0] = (__bf16)a.x; o[1] = (__bf16)a.y; o[2] = (__bf16)a.z; o[3] = (__bf16)a.w;
  o[4] = (__bf16)b.x; o[5] = (__bf16)b.y; o[6] = (__bf16)b.z; o[7] = (__bf16)b.w;
  *(v8bf*)(dst + i) = o;
}

// ---------------------------------------------------------------------------
// Tiled bf16 transpose: dst[c][r] = src[r][c]  (64x64 tiles, per batch in z)
// ---------------------------------------------------------------------------
__global__ __launch_bounds__(256)
void transpose_bf16_kernel(const __bf16* __restrict__ src, __bf16* __restrict__ dst,
                           int R, int Ccols, long long sBatch, long long dBatch) {
  __shared__ alignas(16) __bf16 t[64][80];
  const __bf16* S  = src + (long long)blockIdx.z * sBatch;
  __bf16*       Dp = dst + (long long)blockIdx.z * dBatch;
  const int r0 = blockIdx.y * 64, c0 = blockIdx.x * 64;
  const int tid = threadIdx.x;
#pragma unroll
  for (int i = 0; i < 2; ++i) {
    int ch = tid + i * 256;                 // 512 chunks of 8
    int r = ch >> 3, cc = (ch & 7) * 8;
    v8bf v = *(const v8bf*)(S + (long long)(r0 + r) * Ccols + c0 + cc);
    *(v8bf*)(&t[r][cc]) = v;
  }
  __syncthreads();
#pragma unroll
  for (int i = 0; i < 2; ++i) {
    int ch = tid + i * 256;
    int c = ch >> 3, rr = (ch & 7) * 8;
    v8bf v;
#pragma unroll
    for (int e = 0; e < 8; ++e) v[e] = t[rr + e][c];
    *(v8bf*)(Dp + (long long)(c0 + c) * R + r0 + rr) = v;
  }
}

// ---------------------------------------------------------------------------
// WMMA GEMM:  C[M,N] = alpha * A[M,K] @ B^T + bias,  B memory is [N,K] row-major.
// 256 threads = 8 waves; block tile 256x128; wave tile 64x64 (4x4 accum tiles,
// 16 ds_load_b128 : 16 v_wmma per K-step).  Staging: TDM double-buffered.
// ---------------------------------------------------------------------------
template <typename OutT, bool HAS_BIAS>
__global__ __launch_bounds__(256)
void gemm_wmma_bf16(const __bf16* __restrict__ A, const __bf16* __restrict__ Bm,
                    const float* __restrict__ bias, OutT* __restrict__ C,
                    int N, int Kd,
                    long long strideA, long long strideB, long long strideC,
                    float alpha)
{
  __shared__ alignas(16) __bf16 ldsA[2][TILE_M * LDT];   // 2 x 20 KB
  __shared__ alignas(16) __bf16 ldsB[2][TILE_N * LDT];   // 2 x 10 KB

  const int tid  = threadIdx.x;
  const int lane = tid & 31;
  const int wave = tid >> 5;
  const int wm   = (wave & 3) * 64;   // wave M offset within block tile
  const int wn   = (wave >> 2) * 64;  // wave N offset within block tile
  const int l16  = lane & 15;
  const int half = lane >> 4;

  const int bm = blockIdx.y * TILE_M;
  const int bn = blockIdx.x * TILE_N;

  const __bf16* Atile = A  + (long long)blockIdx.z * strideA + (long long)bm * Kd;
  const __bf16* Btile = Bm + (long long)blockIdx.z * strideB + (long long)bn * Kd;
  OutT*         Cb    = C  + (long long)blockIdx.z * strideC;

  const v8u g1a = tdm_group1(Kd, TILE_M);
  const v8u g1b = tdm_group1(Kd, TILE_N);
  const int ktiles = Kd / TILE_K;

  v8f acc[4][4];
#pragma unroll
  for (int mi = 0; mi < 4; ++mi)
#pragma unroll
    for (int ni = 0; ni < 4; ++ni)
#pragma unroll
      for (int r = 0; r < 8; ++r) acc[mi][ni][r] = 0.0f;

  // prologue: DMA tile 0 into buffer 0
  if (wave == 0) {
    tdm_load_tile(Atile, (unsigned)(size_t)&ldsA[0][0], g1a);
    tdm_load_tile(Btile, (unsigned)(size_t)&ldsB[0][0], g1b);
  }

  for (int kt = 0; kt < ktiles; ++kt) {
    __syncthreads();   // all waves done reading the buffer TDM is about to refill
    if (wave == 0) {
      if (kt + 1 < ktiles) {
        const int nb = (kt + 1) & 1;
        tdm_load_tile(Atile + (kt + 1) * TILE_K, (unsigned)(size_t)&ldsA[nb][0], g1a);
        tdm_load_tile(Btile + (kt + 1) * TILE_K, (unsigned)(size_t)&ldsB[nb][0], g1b);
        __builtin_amdgcn_s_wait_tensorcnt(2);   // tile kt done; kt+1 stays in flight
      } else {
        __builtin_amdgcn_s_wait_tensorcnt(0);
      }
    }
    __syncthreads();   // publish tile kt to all waves
    const int cb = kt & 1;

    // ---- build fragments per ISA layout and issue WMMAs ----
    v16bf afrag[4];
#pragma unroll
    for (int mi = 0; mi < 4; ++mi) {
      const __bf16* p = &ldsA[cb][(wm + mi * 16 + l16) * LDT + half * 8];
      v8bf lo = *(const v8bf*)p;          // K 0..7   (lanes>=16: 8..15)
      v8bf hi = *(const v8bf*)(p + 16);   // K 16..23 (lanes>=16: 24..31)
      afrag[mi] = __builtin_shufflevector(lo, hi, 0,1,2,3,4,5,6,7,8,9,10,11,12,13,14,15);
    }
#pragma unroll
    for (int ni = 0; ni < 4; ++ni) {
      const __bf16* p = &ldsB[cb][(wn + ni * 16 + l16) * LDT + half * 8];
      v8bf lo = *(const v8bf*)p;          // K 0..7   (lanes>=16: 8..15)
      v8bf hi = *(const v8bf*)(p + 16);   // K 16..23 (lanes>=16: 24..31)
      v16bf bfrag = __builtin_shufflevector(lo, hi, 0,1,2,3,4,5,6,7,8,9,10,11,12,13,14,15);
#pragma unroll
      for (int mi = 0; mi < 4; ++mi)
        acc[mi][ni] = __builtin_amdgcn_wmma_f32_16x16x32_bf16(
            false, afrag[mi], false, bfrag, (short)0, acc[mi][ni], false, false);
    }
  }

  // ---- epilogue: C VGPR r -> row (half ? 8+r : r), col = l16 ----
#pragma unroll
  for (int mi = 0; mi < 4; ++mi)
#pragma unroll
    for (int ni = 0; ni < 4; ++ni) {
      int col = bn + wn + ni * 16 + l16;
      float bv = HAS_BIAS ? bias[col] : 0.0f;
#pragma unroll
      for (int r = 0; r < 8; ++r) {
        int row = bm + wm + mi * 16 + (half ? 8 + r : r);
        float v = acc[mi][ni][r] * alpha + bv;
        Cb[(long long)row * N + col] = (OutT)v;
      }
    }
}

// ---------------------------------------------------------------------------
// Row softmax over S=2048 f32 scores -> bf16 probabilities
// ---------------------------------------------------------------------------
__global__ __launch_bounds__(256)
void softmax_rows(const float* __restrict__ Sim, __bf16* __restrict__ P, int S)
{
  __shared__ float red[256];
  long long row = (long long)blockIdx.y * gridDim.x + blockIdx.x;
  const float* src = Sim + row * S;
  __bf16*      dst = P   + row * S;
  int tid = threadIdx.x;

  float vals[8];
  float m = -INFINITY;
#pragma unroll
  for (int i = 0; i < 8; ++i) { vals[i] = src[tid + i * 256]; m = fmaxf(m, vals[i]); }
  red[tid] = m; __syncthreads();
  for (int s = 128; s > 0; s >>= 1) {
    if (tid < s) red[tid] = fmaxf(red[tid], red[tid + s]);
    __syncthreads();
  }
  m = red[0]; __syncthreads();

  float sum = 0.0f;
#pragma unroll
  for (int i = 0; i < 8; ++i) { vals[i] = __expf(vals[i] - m); sum += vals[i]; }
  red[tid] = sum; __syncthreads();
  for (int s = 128; s > 0; s >>= 1) {
    if (tid < s) red[tid] += red[tid + s];
    __syncthreads();
  }
  float inv = 1.0f / red[0];
#pragma unroll
  for (int i = 0; i < 8; ++i) dst[tid + i * 256] = (__bf16)(vals[i] * inv);
}

// ---------------------------------------------------------------------------
// Launcher
// ---------------------------------------------------------------------------
extern "C" void kernel_launch(void* const* d_in, const int* in_sizes, int n_in,
                              void* d_out, int out_size, void* d_ws, size_t ws_size,
                              hipStream_t stream)
{
  (void)in_sizes; (void)n_in; (void)out_size; (void)ws_size;
  const int Bz = 4, S = 2048, D = 1024, M = Bz * S;   // 8192
  const long long SD = (long long)S * D;              // 2,097,152
  const long long SS = (long long)S * S;              // 4,194,304

  const float* X  = (const float*)d_in[0];
  const float* Wq = (const float*)d_in[1];
  const float* bq = (const float*)d_in[2];
  const float* Wk = (const float*)d_in[3];
  const float* bk = (const float*)d_in[4];
  const float* Wv = (const float*)d_in[5];
  const float* bv = (const float*)d_in[6];
  float* out = (float*)d_out;

  char* ws = (char*)d_ws;
  __bf16* Xb   = (__bf16*)(ws);                         // 16 MB
  __bf16* Wqb  = (__bf16*)(ws + 16777216);              //  2 MB
  __bf16* Wkb  = (__bf16*)(ws + 18874368);              //  2 MB
  __bf16* Wvb  = (__bf16*)(ws + 20971520);              //  2 MB
  __bf16* Qb   = (__bf16*)(ws + 23068672);              // 16 MB
  __bf16* Kb   = (__bf16*)(ws + 39845888);              // 16 MB
  __bf16* Vb   = (__bf16*)(ws + 56623104);              // 16 MB
  float*  Sim  = (float*) (ws + 73400320);              // 64 MB
  __bf16* Pb   = (__bf16*)(ws + 140509184);             // 32 MB
  __bf16* Vt   = (__bf16*)(ws + 174063616);             // 16 MB (total ~182 MB)

  // 1) convert inputs to bf16
  f32_to_bf16_kernel<<<(M * D) / 2048, 256, 0, stream>>>(X,  Xb);
  f32_to_bf16_kernel<<<(D * D) / 2048, 256, 0, stream>>>(Wq, Wqb);
  f32_to_bf16_kernel<<<(D * D) / 2048, 256, 0, stream>>>(Wk, Wkb);
  f32_to_bf16_kernel<<<(D * D) / 2048, 256, 0, stream>>>(Wv, Wvb);

  // 2) Q/K/V = X @ W^T + b   (bf16 out)
  dim3 gqkv(D / TILE_N, M / TILE_M, 1);
  gemm_wmma_bf16<__bf16, true><<<gqkv, 256, 0, stream>>>(
      Xb, Wqb, bq, Qb, D, D, 0, 0, 0, 1.0f);
  gemm_wmma_bf16<__bf16, true><<<gqkv, 256, 0, stream>>>(
      Xb, Wkb, bk, Kb, D, D, 0, 0, 0, 1.0f);
  gemm_wmma_bf16<__bf16, true><<<gqkv, 256, 0, stream>>>(
      Xb, Wvb, bv, Vb, D, D, 0, 0, 0, 1.0f);

  // 2b) Vt = V^T per batch (so the PV GEMM uses the same [N,K] TDM path)
  transpose_bf16_kernel<<<dim3(D / 64, S / 64, Bz), 256, 0, stream>>>(
      Vb, Vt, S, D, SD, SD);

  // 3) Sim = (Q @ K^T) / sqrt(D)   per batch (f32, stays in 192MB L2)
  dim3 gsc(S / TILE_N, S / TILE_M, Bz);
  gemm_wmma_bf16<float, false><<<gsc, 256, 0, stream>>>(
      Qb, Kb, nullptr, Sim, S, D, SD, SD, SS, 0.03125f);

  // 4) P = softmax(Sim) rows -> bf16
  softmax_rows<<<dim3(S, Bz), 256, 0, stream>>>(Sim, Pb, S);

  // 5) out = P @ Vt^T per batch (f32 out)
  dim3 gpv(D / TILE_N, S / TILE_M, Bz);
  gemm_wmma_bf16<float, false><<<gpv, 256, 0, stream>>>(
      Pb, Vt, nullptr, out, D, S, SS, SD, SD, 1.0f);
}
